// ResidualAttentionBlock_49933289783809
// MI455X (gfx1250) — compile-verified, matrix operand order checked by reading, not run on hardware
//
#include <hip/hip_runtime.h>
#include <stdint.h>

// ---------------------------------------------------------------------------
// Types
// ---------------------------------------------------------------------------
typedef unsigned short u16;
typedef __attribute__((ext_vector_type(16))) __bf16        v16bf;
typedef __attribute__((ext_vector_type(8)))  float         v8f;
typedef __attribute__((ext_vector_type(4)))  unsigned int  u32x4;
typedef __attribute__((ext_vector_type(4)))  float         f32x4;

#define S_   512
#define B_   32
#define D_   1024
#define Hn   16
#define HD_  64
#define SB   (S_*B_)       // 16384 token rows, row index = s*B + b
#define D3   (3*D_)        // 3072
#define D4   (4*D_)        // 4096

__device__ __forceinline__ u16 f2bf(float f) {
  unsigned u = __float_as_uint(f);
  return (u16)((u + 0x7FFFu + ((u >> 16) & 1u)) >> 16);   // round-to-nearest-even
}

__device__ __forceinline__ v8f vzero() {
  v8f z;
#pragma unroll
  for (int i = 0; i < 8; ++i) z[i] = 0.f;
  return z;
}

union Frag { v16bf v; u32x4 q[2]; };

__device__ __forceinline__ v8f wmma_bf16(v16bf a, v16bf b, v8f c) {
  // D = A(16x32 bf16) x B(32x16 bf16) + C(16x16 f32)
  return __builtin_amdgcn_wmma_f32_16x16x32_bf16(false, a, false, b,
                                                 (short)0, c, false, false);
}

// CDNA5 async global->LDS copy (ASYNCcnt-tracked, no VGPR round trip).
// Flat addresses into LDS use addr[31:0] as the LDS byte offset (ISA 10.2),
// so truncating the generic shared pointer yields the VDST operand.
__device__ __forceinline__ unsigned lds_off(const void* p) {
  return (unsigned)(unsigned long long)p;
}
__device__ __forceinline__ void async_ld_b128(unsigned lds, const void* g) {
  asm volatile("global_load_async_to_lds_b128 %0, %1, off"
               :: "v"(lds), "v"(g) : "memory");
}
__device__ __forceinline__ void wait_async0() {
  asm volatile("s_wait_asynccnt 0x0" ::: "memory");
}

// ---------------------------------------------------------------------------
// K0: zero the attn_weights output region (d_out is poisoned by the harness)
// ---------------------------------------------------------------------------
__global__ __launch_bounds__(256) void zero_f32(float* __restrict__ p) {
  size_t i = ((size_t)blockIdx.x * 256 + threadIdx.x) * 4;
  f32x4 z; z[0] = 0.f; z[1] = 0.f; z[2] = 0.f; z[3] = 0.f;
  *reinterpret_cast<f32x4*>(p + i) = z;
}

// ---------------------------------------------------------------------------
// K1: fp32 -> bf16 weight conversion
// ---------------------------------------------------------------------------
__global__ __launch_bounds__(256) void cvt_bf16(const float* __restrict__ src,
                                                u16* __restrict__ dst) {
  size_t i = ((size_t)blockIdx.x * 256 + threadIdx.x) * 4;
  f32x4 v = *reinterpret_cast<const f32x4*>(src + i);
#pragma unroll
  for (int k = 0; k < 4; ++k) dst[i + k] = f2bf(v[k]);
}

// ---------------------------------------------------------------------------
// K2: LayerNorm over D=1024 per row, fp32 in -> bf16 out
// ---------------------------------------------------------------------------
__global__ __launch_bounds__(256) void ln_bf16(const float* __restrict__ X,
                                               const float* __restrict__ gw,
                                               const float* __restrict__ gb,
                                               u16* __restrict__ Y) {
  __shared__ float red[256];
  const int r = blockIdx.x, tid = threadIdx.x;
  const float* xr = X + (size_t)r * D_;
  f32x4 xv = *reinterpret_cast<const f32x4*>(xr + tid * 4);

  float s = xv[0] + xv[1] + xv[2] + xv[3];
  red[tid] = s; __syncthreads();
  for (int off = 128; off > 0; off >>= 1) {
    if (tid < off) red[tid] += red[tid + off];
    __syncthreads();
  }
  float mu = red[0] * (1.f / D_);
  __syncthreads();

  float d0 = xv[0]-mu, d1 = xv[1]-mu, d2 = xv[2]-mu, d3 = xv[3]-mu;
  red[tid] = d0*d0 + d1*d1 + d2*d2 + d3*d3; __syncthreads();
  for (int off = 128; off > 0; off >>= 1) {
    if (tid < off) red[tid] += red[tid + off];
    __syncthreads();
  }
  float rs = rsqrtf(red[0] * (1.f / D_) + 1e-5f);

  u16* yr = Y + (size_t)r * D_;
#pragma unroll
  for (int k = 0; k < 4; ++k) {
    int c = tid * 4 + k;
    yr[c] = f2bf((xv[k] - mu) * rs * gw[c] + gb[c]);
  }
}

// ---------------------------------------------------------------------------
// K3: bf16 WMMA GEMM  C[m,n] = sum_k A[m,k]*W[n,k] + bias[n]
//     A: (M,K) bf16 row-major, W: (N,K) bf16 row-major (PyTorch Linear layout)
//     mode 0: store bf16(C)        mode 1: float out = res + C
//     mode 2: store bf16(quickGELU(C))
//     Block tile 128M x 128N x 32K, 8 waves (4M x 2N grid, 32x64 per wave,
//     2x4 WMMA accumulators). LDS double-buffered, filled with CDNA5
//     global_load_async_to_lds_b128: tile k+1 streams into buffer ^1 while
//     WMMAs consume buffer cur; only s_wait_asynccnt + barrier per K-step.
// ---------------------------------------------------------------------------
__global__ __launch_bounds__(256) void gemm_bf16(
    const u16* __restrict__ A, const u16* __restrict__ W,
    const float* __restrict__ bias, const float* __restrict__ res,
    void* __restrict__ Cout, int M, int N, int K, int mode) {
  __shared__ __align__(16) u16 As[2][128 * 40];   // 32-elem rows padded to 40
  __shared__ __align__(16) u16 Bs[2][128 * 40];

  const int tid  = threadIdx.x;
  const int lane = tid & 31, wave = tid >> 5;
  const int wm = wave & 3, wn = wave >> 2;        // 4(M) x 2(N) wave grid
  const int m0 = blockIdx.y * 128;
  const int n0 = blockIdx.x * 128;
  const int row = lane & 15, kb = (lane >> 4) * 8;

  // staging: 128x32 tile = 512 16B chunks = 2 per thread (for A and for W)
  const int sr = tid >> 2, cc = tid & 3;
  const int soff = sr * 40 + cc * 8;              // elements into LDS tile
  const u16* gA0 = A + (size_t)(m0 + sr) * K + cc * 8;
  const u16* gA1 = A + (size_t)(m0 + sr + 64) * K + cc * 8;
  const u16* gB0 = W + (size_t)(n0 + sr) * K + cc * 8;
  const u16* gB1 = W + (size_t)(n0 + sr + 64) * K + cc * 8;

  v8f acc[2][4];
#pragma unroll
  for (int i = 0; i < 2; ++i)
#pragma unroll
    for (int j = 0; j < 4; ++j) acc[i][j] = vzero();

  // prologue: async-prefetch tile 0 into buffer 0
  async_ld_b128(lds_off(&As[0][soff]),            gA0);
  async_ld_b128(lds_off(&As[0][soff + 64 * 40]),  gA1);
  async_ld_b128(lds_off(&Bs[0][soff]),            gB0);
  async_ld_b128(lds_off(&Bs[0][soff + 64 * 40]),  gB1);

  for (int k0 = 0; k0 < K; k0 += 32) {
    const int cur = (k0 >> 5) & 1;
    wait_async0();          // this wave's tile-cur copies landed in LDS
    __syncthreads();        // ... and every other wave's too; prev reads done

    if (k0 + 32 < K) {      // stream tile k+1 into the other buffer
      const int nxt = cur ^ 1;
      const int ko  = k0 + 32;
      async_ld_b128(lds_off(&As[nxt][soff]),           gA0 + ko);
      async_ld_b128(lds_off(&As[nxt][soff + 64 * 40]), gA1 + ko);
      async_ld_b128(lds_off(&Bs[nxt][soff]),           gB0 + ko);
      async_ld_b128(lds_off(&Bs[nxt][soff + 64 * 40]), gB1 + ko);
    }

    Frag af[2], bf[4];
#pragma unroll
    for (int mt = 0; mt < 2; ++mt) {
      const u16* p = &As[cur][(wm * 32 + mt * 16 + row) * 40 + kb];
      af[mt].q[0] = *reinterpret_cast<const u32x4*>(p);
      af[mt].q[1] = *reinterpret_cast<const u32x4*>(p + 16);
    }
#pragma unroll
    for (int nt = 0; nt < 4; ++nt) {
      const u16* p = &Bs[cur][(wn * 64 + nt * 16 + row) * 40 + kb];
      bf[nt].q[0] = *reinterpret_cast<const u32x4*>(p);
      bf[nt].q[1] = *reinterpret_cast<const u32x4*>(p + 16);
    }
#pragma unroll
    for (int mt = 0; mt < 2; ++mt)
#pragma unroll
      for (int nt = 0; nt < 4; ++nt)
        acc[mt][nt] = wmma_bf16(af[mt].v, bf[nt].v, acc[mt][nt]);
  }

  // epilogue: C element (VGPR v, lane l) -> M = v + 8*(l>>4), N = l&15
#pragma unroll
  for (int mt = 0; mt < 2; ++mt)
#pragma unroll
    for (int nt = 0; nt < 4; ++nt) {
      int n = n0 + wn * 64 + nt * 16 + (lane & 15);
      float bv = bias[n];
#pragma unroll
      for (int v = 0; v < 8; ++v) {
        int m = m0 + wm * 32 + mt * 16 + v + ((lane >> 4) << 3);
        float val = acc[mt][nt][v] + bv;
        size_t ci = (size_t)m * N + n;
        if (mode == 0) {
          ((u16*)Cout)[ci] = f2bf(val);
        } else if (mode == 1) {
          ((float*)Cout)[ci] = res[ci] + val;
        } else {
          float g = val / (1.f + __expf(-1.702f * val));   // quickGELU
          ((u16*)Cout)[ci] = f2bf(g);
        }
      }
    }
}

// ---------------------------------------------------------------------------
// K4: build V^T (B,H,HD,S) bf16 from qkv (SB,3D) bf16 so the P x V
//     B-fragment is K-contiguous (key-position contiguous per hd lane).
// ---------------------------------------------------------------------------
__global__ __launch_bounds__(256) void vtrans(const u16* __restrict__ qkv,
                                              u16* __restrict__ vT) {
  size_t base = ((size_t)blockIdx.x * 256 + threadIdx.x) * 4;
#pragma unroll
  for (int k = 0; k < 4; ++k) {
    size_t i  = base + k;
    int s     = (int)(i & (S_ - 1));
    size_t r  = i >> 9;
    int hd    = (int)(r & (HD_ - 1));
    size_t r2 = r >> 6;
    int h     = (int)(r2 & (Hn - 1));
    int b     = (int)(r2 >> 4);
    vT[i] = qkv[(size_t)(s * B_ + b) * D3 + 2 * D_ + h * HD_ + hd];
  }
}

// ---------------------------------------------------------------------------
// K5: fused attention. Block = (query-tile of 64 s-values, batch b),
//     8 waves, loops over all 16 heads. Per head:
//       P(64x512) = (Q K^T)/8 via WMMA  -> LDS fp32
//       softmax rows (4 threads/row)
//       attn_weights[b,q,:] += P/16     (block owns rows, no atomics)
//       ctx(64x64) = P x V via WMMA (P re-quantized to bf16 in LDS)
//     Dynamic LDS: P 64x520 f32 + Pb 64x520 bf16 + reduce scratch ~ 200 KB
// ---------------------------------------------------------------------------
#define PSTR 520
#define ATTN_SMEM (64*PSTR*4 + 64*PSTR*2 + 256*4)

__global__ __launch_bounds__(256) void attn_kernel(
    const u16* __restrict__ qkv, const u16* __restrict__ vT,
    float* __restrict__ attnw, u16* __restrict__ ctx) {
  extern __shared__ __align__(16) char smem[];
  float* P   = reinterpret_cast<float*>(smem);
  u16*   Pb  = reinterpret_cast<u16*>(smem + 64 * PSTR * 4);
  float* red = reinterpret_cast<float*>(smem + 64 * PSTR * 4 + 64 * PSTR * 2);

  const int tid  = threadIdx.x;
  const int lane = tid & 31, wave = tid >> 5;
  const int q0   = blockIdx.x * 64;
  const int b    = blockIdx.y;
  const int row  = lane & 15, kb = (lane >> 4) * 8;

  for (int h = 0; h < Hn; ++h) {
    __syncthreads();   // previous head's Pb consumers are done

    // ---- phase 1: scores. Wave covers key columns [wave*64, wave*64+64)
    v8f acc[4][4];
#pragma unroll
    for (int i = 0; i < 4; ++i)
#pragma unroll
      for (int j = 0; j < 4; ++j) acc[i][j] = vzero();

#pragma unroll
    for (int ks = 0; ks < 2; ++ks) {          // HD = 64 = 2 x 32
      Frag qa[4], kf[4];
#pragma unroll
      for (int mt = 0; mt < 4; ++mt) {
        int srow = q0 + mt * 16 + row;
        const u16* p = qkv + (size_t)(srow * B_ + b) * D3 + h * HD_ + ks * 32 + kb;
        qa[mt].q[0] = *reinterpret_cast<const u32x4*>(p);
        qa[mt].q[1] = *reinterpret_cast<const u32x4*>(p + 16);
      }
#pragma unroll
      for (int nt = 0; nt < 4; ++nt) {
        int krow = wave * 64 + nt * 16 + row;
        const u16* p = qkv + (size_t)(krow * B_ + b) * D3 + D_ + h * HD_ + ks * 32 + kb;
        kf[nt].q[0] = *reinterpret_cast<const u32x4*>(p);
        kf[nt].q[1] = *reinterpret_cast<const u32x4*>(p + 16);
      }
#pragma unroll
      for (int mt = 0; mt < 4; ++mt)
#pragma unroll
        for (int nt = 0; nt < 4; ++nt)
          acc[mt][nt] = wmma_bf16(qa[mt].v, kf[nt].v, acc[mt][nt]);
    }
#pragma unroll
    for (int mt = 0; mt < 4; ++mt)
#pragma unroll
      for (int nt = 0; nt < 4; ++nt)
#pragma unroll
        for (int v = 0; v < 8; ++v) {
          int r = mt * 16 + v + ((lane >> 4) << 3);
          int c = wave * 64 + nt * 16 + (lane & 15);
          P[r * PSTR + c] = acc[mt][nt][v] * 0.125f;   // 1/sqrt(64)
        }
    __syncthreads();

    // ---- phase 2: softmax. 4 threads per row, 128 cols each.
    const int r2 = tid >> 2, qtr = tid & 3, cb = qtr * 128;
    float mx = -3.0e38f;
    for (int j = 0; j < 128; ++j) mx = fmaxf(mx, P[r2 * PSTR + cb + j]);
    red[r2 * 4 + qtr] = mx; __syncthreads();
    float rmax = fmaxf(fmaxf(red[r2 * 4], red[r2 * 4 + 1]),
                       fmaxf(red[r2 * 4 + 2], red[r2 * 4 + 3]));
    __syncthreads();
    float sum = 0.f;
    for (int j = 0; j < 128; ++j) {
      float e = __expf(P[r2 * PSTR + cb + j] - rmax);
      P[r2 * PSTR + cb + j] = e;
      sum += e;
    }
    red[r2 * 4 + qtr] = sum; __syncthreads();
    float inv = 1.f / (red[r2 * 4] + red[r2 * 4 + 1] +
                       red[r2 * 4 + 2] + red[r2 * 4 + 3]);
    float* aw = attnw + ((size_t)b * S_ + q0 + r2) * S_;
    for (int j = 0; j < 128; ++j) {
      int c = cb + j;
      float pn = P[r2 * PSTR + c] * inv;
      Pb[r2 * PSTR + c] = f2bf(pn);
      aw[c] += pn * (1.f / Hn);     // head-mean; block owns these rows
    }
    __syncthreads();

    // ---- phase 3: ctx = P x V. 16 tiles (4M x 4N) over 8 waves.
    const int mt  = wave & 3;
    const int nt0 = (wave >> 2) * 2;
    v8f c0 = vzero(), c1 = vzero();
    for (int ks = 0; ks < 16; ++ks) {          // S = 512 = 16 x 32
      Frag pa;
      const u16* pp = Pb + (mt * 16 + row) * PSTR + ks * 32 + kb;
      pa.q[0] = *reinterpret_cast<const u32x4*>(pp);
      pa.q[1] = *reinterpret_cast<const u32x4*>(pp + 16);
#pragma unroll
      for (int t = 0; t < 2; ++t) {
        int hd = (nt0 + t) * 16 + row;
        const u16* vp = vT + ((size_t)(b * Hn + h) * HD_ + hd) * S_ + ks * 32 + kb;
        Frag vf;
        vf.q[0] = *reinterpret_cast<const u32x4*>(vp);
        vf.q[1] = *reinterpret_cast<const u32x4*>(vp + 16);
        if (t == 0) c0 = wmma_bf16(pa.v, vf.v, c0);
        else        c1 = wmma_bf16(pa.v, vf.v, c1);
      }
    }
#pragma unroll
    for (int t = 0; t < 2; ++t) {
      const v8f& cc = t ? c1 : c0;
#pragma unroll
      for (int v = 0; v < 8; ++v) {
        int r = mt * 16 + v + ((lane >> 4) << 3);
        int s = q0 + r;
        int col = h * HD_ + (nt0 + t) * 16 + (lane & 15);
        ctx[(size_t)(s * B_ + b) * D_ + col] = f2bf(cc[v]);
      }
    }
  }
}

// ---------------------------------------------------------------------------
// Launcher
// ---------------------------------------------------------------------------
extern "C" void kernel_launch(void* const* d_in, const int* in_sizes, int n_in,
                              void* d_out, int out_size, void* d_ws, size_t ws_size,
                              hipStream_t stream) {
  const float* x         = (const float*)d_in[0];
  const float* ln1_w     = (const float*)d_in[1];
  const float* ln1_b     = (const float*)d_in[2];
  const float* in_proj_w = (const float*)d_in[3];
  const float* in_proj_b = (const float*)d_in[4];
  const float* out_w     = (const float*)d_in[5];
  const float* out_b     = (const float*)d_in[6];
  const float* ln2_w     = (const float*)d_in[7];
  const float* ln2_b     = (const float*)d_in[8];
  const float* fc_w      = (const float*)d_in[9];
  const float* fc_b      = (const float*)d_in[10];
  const float* proj_w    = (const float*)d_in[11];
  const float* proj_b    = (const float*)d_in[12];

  float* out_x    = (float*)d_out;                        // (S,B,D)
  float* out_attn = out_x + (size_t)SB * D_;              // (B,S,S)

  char* ws = (char*)d_ws;
  u16*   WQb  = (u16*)  (ws + 0);                         //  6,291,456 B
  u16*   WOb  = (u16*)  (ws + 6291456);                   //  2,097,152 B
  u16*   WFCb = (u16*)  (ws + 8388608);                   //  8,388,608 B
  u16*   WPRb = (u16*)  (ws + 16777216);                  //  8,388,608 B
  u16*   Hb   = (u16*)  (ws + 25165824);                  // 33,554,432 B (LN1/LN2 out)
  u16*   QKVb = (u16*)  (ws + 58720256);                  // 100,663,296 B
  u16*   VTb  = (u16*)  (ws + 159383552);                 // 33,554,432 B
  u16*   CTXb = (u16*)  (ws + 192937984);                 // 33,554,432 B
  float* X2   = (float*)(ws + 226492416);                 // 67,108,864 B
  u16*   Gb   = QKVb;   // GELU output (SB x 4D bf16) overlays dead QKV+VT

  hipFuncSetAttribute((const void*)attn_kernel,
                      hipFuncAttributeMaxDynamicSharedMemorySize, ATTN_SMEM);

  // attn_weights region must start at zero (harness poisons d_out)
  zero_f32<<<dim3((B_ * S_ * S_) / 1024), dim3(256), 0, stream>>>(out_attn);

  // weights -> bf16
  cvt_bf16<<<dim3((D3 * D_) / 1024), dim3(256), 0, stream>>>(in_proj_w, WQb);
  cvt_bf16<<<dim3((D_ * D_) / 1024), dim3(256), 0, stream>>>(out_w, WOb);
  cvt_bf16<<<dim3((D4 * D_) / 1024), dim3(256), 0, stream>>>(fc_w, WFCb);
  cvt_bf16<<<dim3((D_ * D4) / 1024), dim3(256), 0, stream>>>(proj_w, WPRb);

  // h = LN1(x) -> bf16
  ln_bf16<<<dim3(SB), dim3(256), 0, stream>>>(x, ln1_w, ln1_b, Hb);

  // qkv = h @ in_proj_w^T + b   (SB x 3072) bf16
  gemm_bf16<<<dim3(D3 / 128, SB / 128), dim3(256), 0, stream>>>(
      Hb, WQb, in_proj_b, nullptr, (void*)QKVb, SB, D3, D_, 0);

  // V^T rearrange for P x V
  vtrans<<<dim3((SB * D_) / 1024), dim3(256), 0, stream>>>(QKVb, VTb);

  // fused attention: ctx bf16 + head-mean attn weights
  attn_kernel<<<dim3(S_ / 64, B_), dim3(256), ATTN_SMEM, stream>>>(
      QKVb, VTb, out_attn, CTXb);

  // x2 = x + ctx @ out_w^T + b
  gemm_bf16<<<dim3(D_ / 128, SB / 128), dim3(256), 0, stream>>>(
      CTXb, WOb, out_b, x, (void*)X2, SB, D_, D_, 1);

  // m = LN2(x2) -> bf16 (reuses Hb)
  ln_bf16<<<dim3(SB), dim3(256), 0, stream>>>(X2, ln2_w, ln2_b, Hb);

  // g = quickGELU(m @ fc_w^T + b) -> bf16 (SB x 4096)
  gemm_bf16<<<dim3(D4 / 128, SB / 128), dim3(256), 0, stream>>>(
      Hb, WFCb, fc_b, nullptr, (void*)Gb, SB, D4, D_, 2);

  // out = x2 + g @ proj_w^T + b
  gemm_bf16<<<dim3(D_ / 128, SB / 128), dim3(256), 0, stream>>>(
      Gb, WPRb, proj_b, X2, (void*)out_x, SB, D_, D4, 1);
}
